// AnchorGNNPocket_85289460564116
// MI455X (gfx1250) — matrix operand order repaired and checked by reading, not compiled, not run on hardware
//
#include <hip/hip_runtime.h>
#include <hip/hip_bf16.h>

// ---------------------------------------------------------------------------
// AnchorGNNPocket on CDNA5 (gfx1250): fused edge-MLP via v_wmma_f32_16x16x32_f16
// B=8, NS=32, NP=224, N=256, HID=128, NLAYERS=4
// ---------------------------------------------------------------------------

#define BB      8
#define NS_     32
#define NP_     224
#define NN      256
#define HID_    128
#define CUT2_   20.25f
#define INORM   0.01f

typedef __attribute__((ext_vector_type(16))) _Float16 v16h;
typedef __attribute__((ext_vector_type(8)))  float    v8f;

union V16H { v16h v; _Float16 h[16]; };
union V8F  { v8f  v; float    f[8];  };

// ---------------------------------------------------------------------------
// Embedding: h = cat(silu(mol_h@W_mol+b_mol), silu(pok_h@W_pok+b_pok)) @ W_emb + b_emb
// ---------------------------------------------------------------------------
__global__ __launch_bounds__(128) void embed_kernel(
    const float* __restrict__ mol_h, const float* __restrict__ pocket_h,
    const float* __restrict__ W_mol, const float* __restrict__ b_mol,
    const float* __restrict__ W_pok, const float* __restrict__ b_pok,
    const float* __restrict__ W_emb, const float* __restrict__ b_emb,
    float* __restrict__ h)
{
    int row = blockIdx.x;              // b*256 + n
    int b = row >> 8, n = row & 255;
    int c = threadIdx.x;
    __shared__ float sJ[32];
    if (c < 32) {
        float acc;
        if (n < NS_) {
            acc = b_mol[c];
            const float* r = mol_h + (b * NS_ + n) * 10;
            #pragma unroll
            for (int k = 0; k < 10; ++k) acc += r[k] * W_mol[k * 32 + c];
        } else {
            acc = b_pok[c];
            const float* r = pocket_h + (b * NP_ + (n - NS_)) * 25;
            #pragma unroll
            for (int k = 0; k < 25; ++k) acc += r[k] * W_pok[k * 32 + c];
        }
        sJ[c] = acc / (1.f + __expf(-acc));      // silu
    }
    __syncthreads();
    float acc = b_emb[c];
    #pragma unroll 8
    for (int k = 0; k < 32; ++k) acc += sJ[k] * W_emb[k * HID_ + c];
    h[row * HID_ + c] = acc;
}

// ---------------------------------------------------------------------------
// Geometry: d2[b,i,j] and masked adjacency weight wadj[b,i,j]
// ---------------------------------------------------------------------------
__global__ __launch_bounds__(256) void geom_kernel(
    const float* __restrict__ mol_x, const float* __restrict__ pocket_x,
    const float* __restrict__ node_mask, const float* __restrict__ pocket_mask,
    float* __restrict__ d2, float* __restrict__ wadj)
{
    int bi = blockIdx.x;               // b*256 + i
    int b = bi >> 8, i = bi & 255;
    int j = threadIdx.x;
    __shared__ float xi[3];
    __shared__ float mi;
    auto getx = [&](int n, int d) -> float {
        return (n < NS_) ? mol_x[(b * NS_ + n) * 3 + d]
                         : pocket_x[(b * NP_ + (n - NS_)) * 3 + d];
    };
    auto getm = [&](int n) -> float {
        return (n < NS_) ? node_mask[b * NS_ + n] : pocket_mask[b * NP_ + (n - NS_)];
    };
    if (j < 3) xi[j] = getx(i, j);
    if (j == 3) mi = getm(i);
    __syncthreads();
    float dx = getx(j, 0) - xi[0];
    float dy = getx(j, 1) - xi[1];
    float dz = getx(j, 2) - xi[2];
    float dd = dx * dx + dy * dy + dz * dz;
    d2[bi * NN + j] = dd;
    float lig = ((i < NS_) && (j < NS_)) ? 1.f : ((dd <= CUT2_) ? 1.f : 0.f);
    wadj[bi * NN + j] = lig * mi * getm(j);
}

// ---------------------------------------------------------------------------
// Generic WMMA GEMM: D(rows x 128) = op( [A1|A2](rows x K) @ Bm(K x 128) + bias ) + resid
// Block = 256 threads (8 waves), one 16x128 output tile per block.
// ---------------------------------------------------------------------------
__global__ __launch_bounds__(256) void gemm_wmma(
    const float* __restrict__ A1, const float* __restrict__ A2,
    int k1, int K,
    const float* __restrict__ Bm,
    const float* __restrict__ bias,
    const float* __restrict__ resid,
    float* __restrict__ D,
    int do_relu)
{
    __shared__ _Float16 sA[16][40];            // 16 rows x 32 K (f16), padded
    int tid  = threadIdx.x;
    int lane = tid & 31, wave = tid >> 5;
    int rtile = blockIdx.x;
    int col   = wave * 16 + (lane & 15);

    V8F c; c.v = (v8f){};
    int arow  = lane & 15;
    int abase = (lane < 16) ? 0 : 8;
    int kb0   = (lane < 16) ? 0 : 16;

    for (int kb = 0; kb < K; kb += 32) {
        // cooperative A-tile load (f32 -> f16), 2 elements/thread
        int r  = tid >> 4;
        int kk = (tid & 15) * 2;
        int k  = kb + kk;
        int rg = rtile * 16 + r;
        float x0, x1;
        if (k < k1) {
            const float* p = A1 + (size_t)rg * k1 + k;
            x0 = p[0]; x1 = p[1];
        } else {
            const float* p = A2 + (size_t)rg * (K - k1) + (k - k1);
            x0 = p[0]; x1 = p[1];
        }
        sA[r][kk]     = (_Float16)x0;
        sA[r][kk + 1] = (_Float16)x1;
        __syncthreads();

        // B fragment straight from global (row-major K x 128)
        V16H bf;
        #pragma unroll
        for (int e = 0; e < 16; ++e)
            bf.h[e] = (_Float16)Bm[(size_t)(kb + (kb0 >> 1) + e + (e >= 8 ? 8 : 0)) * HID_ + col];

        // A fragment from LDS per ISA 16-bit A 16x32 layout
        V16H a;
        #pragma unroll
        for (int e = 0; e < 8; ++e)  a.h[e] = sA[arow][abase + e];
        #pragma unroll
        for (int e = 8; e < 16; ++e) a.h[e] = sA[arow][abase + e + 8];

        c.v = __builtin_amdgcn_wmma_f32_16x16x32_f16(false, a.v, false, bf.v,
                                                     (short)0, c.v, false, false);
        __syncthreads();
    }

    int rbase = (lane < 16) ? 0 : 8;
    float bb = bias ? bias[col] : 0.f;
    #pragma unroll
    for (int v = 0; v < 8; ++v) {
        int rg = rtile * 16 + rbase + v;
        float val = c.f[v] + bb;
        if (do_relu) val = fmaxf(val, 0.f);
        if (resid)   val += resid[(size_t)rg * HID_ + col];
        D[(size_t)rg * HID_ + col] = val;
    }
}

// ---------------------------------------------------------------------------
// Fused edge MLP + attention gate + adjacency aggregation.
// One block per (b,i): 8 waves, wave w owns h-columns [16w,16w+16).
// Per 16-j tile: build relu(pre) (16x128 f16) in LDS, 4 WMMAs/wave vs
// register-resident We2 fragments; m stays in registers for the aggregation,
// LDS sM only feeds the sigmoid(m@Wat) gate reduction. Tiles whose adjacency
// row is entirely zero are skipped block-uniformly (gate == 0 exactly).
// ---------------------------------------------------------------------------
__global__ __launch_bounds__(256) void edge_kernel(
    const float* __restrict__ ha,    // (B,256,128) h@wa
    const float* __restrict__ hb,    // (B,256,128) h@wb
    const float* __restrict__ d2,    // (B,256,256)
    const float* __restrict__ wadj,  // (B,256,256)
    const float* __restrict__ wc,    // (128) = We1[l][2H]
    const float* __restrict__ be1,   // (128)
    const float* __restrict__ We2,   // (128,128) K x H row-major
    const float* __restrict__ be2,   // (128)
    const float* __restrict__ Wat,   // (128)
    const float* __restrict__ batp,  // (1)
    float* __restrict__ agg)         // (B,256,128)
{
    __shared__ _Float16 sA[16][136];   // relu(pre) tile, f16, padded rows
    __shared__ float    sM[16][132];   // m tile, f32 (gate reduction only)
    __shared__ float    sBase[HID_];   // ha[b,i,:] + be1
    __shared__ float    sWc[HID_];
    __shared__ float    sWat[HID_];
    __shared__ float    sW[NN];        // wadj row for this (b,i)
    __shared__ float    sD2[NN];       // d2 row for this (b,i)
    __shared__ int      sNz[16];       // per-j-tile nonzero flags
    __shared__ float    sRed[16][17];
    __shared__ float    sGate[16];
    __shared__ float    sAcc[256];

    int tid  = threadIdx.x;
    int lane = tid & 31, wave = tid >> 5;
    int b = blockIdx.x >> 8;
    int i = blockIdx.x & 255;
    size_t bi = (size_t)b * NN + i;

    if (tid < HID_) {
        sBase[tid] = ha[bi * HID_ + tid] + be1[tid];
        sWc[tid]   = wc[tid];
        sWat[tid]  = Wat[tid];
    }
    if (tid < 16) sNz[tid] = 0;
    __syncthreads();
    {
        float w = wadj[bi * NN + tid];
        sW[tid]  = w;
        sD2[tid] = d2[bi * NN + tid];
        if (w != 0.f) atomicOr(&sNz[tid >> 4], 1);
    }

    // We2 B-fragments live in registers for the whole kernel (4 K-steps)
    int col = wave * 16 + (lane & 15);
    int kb0 = (lane < 16) ? 0 : 16;
    V16H bfrag[4];
    #pragma unroll
    for (int ks = 0; ks < 4; ++ks)
        #pragma unroll
        for (int e = 0; e < 16; ++e)
            bfrag[ks].h[e] = (_Float16)We2[(size_t)(ks * 32 + (kb0 >> 1) + e + (e >= 8 ? 8 : 0)) * HID_ + col];

    float batv = batp[0];
    int jr   = tid >> 4;            // row within j-tile this thread services
    int k0   = (tid & 15) * 8;      // k-range this thread services
    int arow  = lane & 15;
    int asel  = (lane < 16) ? 0 : 8;
    int rbase = (lane < 16) ? 0 : 8;
    float bb  = be2[col];
    float accCol = 0.f;             // agg partial: this lane's column, 8 rows
    __syncthreads();

    for (int jt = 0; jt < 16; ++jt) {
        if (!sNz[jt]) continue;     // block-uniform skip: whole tile has gate==0

        // ---- build relu(pre) tile in LDS (f16): pre = base + hb[j] + d2*wc ----
        int jg = jt * 16 + jr;
        float d2v = sD2[jg];
        const float* hbrow = hb + ((size_t)b * NN + jg) * HID_;
        #pragma unroll
        for (int q = 0; q < 8; ++q) {
            int k = k0 + q;
            float pre = sBase[k] + hbrow[k] + d2v * sWc[k];
            sA[jr][k] = (_Float16)fmaxf(pre, 0.f);
        }
        __syncthreads();

        // ---- 4x WMMA: (16j x 128k) @ We2(128k x 16h) ----
        V8F c; c.v = (v8f){};
        #pragma unroll
        for (int ks = 0; ks < 4; ++ks) {
            V16H a;
            int base = ks * 32 + asel;
            #pragma unroll
            for (int e = 0; e < 8; ++e)  a.h[e] = sA[arow][base + e];
            #pragma unroll
            for (int e = 8; e < 16; ++e) a.h[e] = sA[arow][base + e + 8];
            c.v = __builtin_amdgcn_wmma_f32_16x16x32_f16(false, a.v, false, bfrag[ks].v,
                                                         (short)0, c.v, false, false);
        }
        // m = relu(c + be2): keep in registers, mirror to LDS for the gate dot
        #pragma unroll
        for (int v = 0; v < 8; ++v) {
            c.f[v] = fmaxf(c.f[v] + bb, 0.f);
            sM[rbase + v][col] = c.f[v];
        }
        __syncthreads();

        // ---- gate: sigmoid(m @ Wat + bat), fused with adjacency weight ----
        {
            int s = tid & 15;
            float part = 0.f;
            #pragma unroll
            for (int q = 0; q < 8; ++q) part += sM[jr][s + 16 * q] * sWat[s + 16 * q];
            sRed[jr][s] = part;
        }
        __syncthreads();
        if (tid < 16) {
            float dot = 0.f;
            #pragma unroll
            for (int s = 0; s < 16; ++s) dot += sRed[tid][s];
            float w = sW[jt * 16 + tid];
            sGate[tid] = w / (1.f + __expf(-(dot + batv)));
        }
        __syncthreads();

        // ---- agg accumulation straight from the WMMA registers ----
        #pragma unroll
        for (int v = 0; v < 8; ++v)
            accCol += sGate[rbase + v] * c.f[v];
        // no trailing barrier needed: sGate/sA/sM rewrites are fenced by the
        // three barriers of the next executed tile
    }

    __syncthreads();
    sAcc[tid] = accCol;
    __syncthreads();
    if (lane < 16)
        agg[bi * HID_ + col] = (sAcc[tid] + sAcc[tid + 16]) * INORM;
}

// ---------------------------------------------------------------------------
// Output head: out = (relu(h_lig @ W_out + b_out) @ W_lin + b_lin) * mask
// ---------------------------------------------------------------------------
__global__ __launch_bounds__(128) void out_kernel(
    const float* __restrict__ h, const float* __restrict__ node_mask,
    const float* __restrict__ W_out, const float* __restrict__ b_out,
    const float* __restrict__ W_lin, const float* __restrict__ b_lin,
    float* __restrict__ out)
{
    int row = blockIdx.x;              // b*32 + s
    int b = row >> 5, s = row & 31;
    int c = threadIdx.x;
    __shared__ float sH[HID_];
    __shared__ float sR[HID_];
    sH[c] = h[((size_t)b * NN + s) * HID_ + c];
    __syncthreads();
    float acc = b_out[c];
    #pragma unroll 8
    for (int k = 0; k < HID_; ++k) acc += sH[k] * W_out[k * HID_ + c];
    sR[c] = fmaxf(acc, 0.f) * W_lin[c];
    __syncthreads();
    for (int stride = 64; stride > 0; stride >>= 1) {
        if (c < stride) sR[c] += sR[c + stride];
        __syncthreads();
    }
    if (c == 0) out[row] = (sR[0] + b_lin[0]) * node_mask[b * NS_ + s];
}

// ---------------------------------------------------------------------------
extern "C" void kernel_launch(void* const* d_in, const int* in_sizes, int n_in,
                              void* d_out, int out_size, void* d_ws, size_t ws_size,
                              hipStream_t stream)
{
    (void)in_sizes; (void)n_in; (void)out_size; (void)ws_size;
    const float* mol_x      = (const float*)d_in[0];
    const float* mol_h      = (const float*)d_in[1];
    const float* node_mask  = (const float*)d_in[2];
    const float* pocket_x   = (const float*)d_in[3];
    const float* pocket_h   = (const float*)d_in[4];
    const float* pocket_mask= (const float*)d_in[5];
    const float* W_mol      = (const float*)d_in[6];
    const float* b_mol      = (const float*)d_in[7];
    const float* W_pok      = (const float*)d_in[8];
    const float* b_pok      = (const float*)d_in[9];
    const float* W_emb      = (const float*)d_in[10];
    const float* b_emb      = (const float*)d_in[11];
    const float* We1        = (const float*)d_in[12]; // (4,257,128)
    const float* be1        = (const float*)d_in[13]; // (4,128)
    const float* We2        = (const float*)d_in[14]; // (4,128,128)
    const float* be2        = (const float*)d_in[15]; // (4,128)
    const float* Wat        = (const float*)d_in[16]; // (4,128,1)
    const float* bat        = (const float*)d_in[17]; // (4,1)
    const float* Wn1        = (const float*)d_in[18]; // (4,256,128)
    const float* bn1        = (const float*)d_in[19]; // (4,128)
    const float* Wn2        = (const float*)d_in[20]; // (4,128,128)
    const float* bn2        = (const float*)d_in[21]; // (4,128)
    const float* W_out      = (const float*)d_in[22]; // (128,128)
    const float* b_out      = (const float*)d_in[23];
    const float* W_lin      = (const float*)d_in[24]; // (128,1)
    const float* b_lin      = (const float*)d_in[25];

    const int ROWS = BB * NN;                       // 2048
    char* ws = (char*)d_ws;
    float* h    = (float*)ws; ws += (size_t)ROWS * HID_ * 4;     // 1 MB
    float* ha   = (float*)ws; ws += (size_t)ROWS * HID_ * 4;     // 1 MB
    float* hbuf = (float*)ws; ws += (size_t)ROWS * HID_ * 4;     // 1 MB
    float* agg  = (float*)ws; ws += (size_t)ROWS * HID_ * 4;     // 1 MB
    float* t1   = (float*)ws; ws += (size_t)ROWS * HID_ * 4;     // 1 MB
    float* d2   = (float*)ws; ws += (size_t)BB * NN * NN * 4;    // 2 MB
    float* wadj = (float*)ws; ws += (size_t)BB * NN * NN * 4;    // 2 MB

    embed_kernel<<<ROWS, 128, 0, stream>>>(mol_h, pocket_h, W_mol, b_mol,
                                           W_pok, b_pok, W_emb, b_emb, h);
    geom_kernel<<<ROWS, 256, 0, stream>>>(mol_x, pocket_x, node_mask, pocket_mask,
                                          d2, wadj);

    for (int l = 0; l < 4; ++l) {
        const float* We1l = We1 + (size_t)l * 257 * HID_;
        const float* wa = We1l;                     // rows 0..127
        const float* wb = We1l + 128 * HID_;        // rows 128..255
        const float* wc = We1l + 256 * HID_;        // row 256

        // ha = h @ wa ; hb = h @ wb
        gemm_wmma<<<ROWS / 16, 256, 0, stream>>>(h, nullptr, HID_, HID_, wa,
                                                 nullptr, nullptr, ha, 0);
        gemm_wmma<<<ROWS / 16, 256, 0, stream>>>(h, nullptr, HID_, HID_, wb,
                                                 nullptr, nullptr, hbuf, 0);

        // fused edge MLP + gate + aggregation
        edge_kernel<<<ROWS, 256, 0, stream>>>(ha, hbuf, d2, wadj, wc,
                                              be1 + l * HID_,
                                              We2 + (size_t)l * HID_ * HID_,
                                              be2 + l * HID_,
                                              Wat + l * HID_,
                                              bat + l,
                                              agg);

        // node update: t1 = relu(cat(h,agg) @ Wn1 + bn1); h += t1 @ Wn2 + bn2
        gemm_wmma<<<ROWS / 16, 256, 0, stream>>>(h, agg, HID_, 2 * HID_,
                                                 Wn1 + (size_t)l * 2 * HID_ * HID_,
                                                 bn1 + l * HID_, nullptr, t1, 1);
        gemm_wmma<<<ROWS / 16, 256, 0, stream>>>(t1, nullptr, HID_, HID_,
                                                 Wn2 + (size_t)l * HID_ * HID_,
                                                 bn2 + l * HID_, h, h, 0);
    }

    out_kernel<<<BB * NS_, 128, 0, stream>>>(h, node_mask, W_out, b_out,
                                             W_lin, b_lin, (float*)d_out);
}
